// VaryMambaModel_88905823027934
// MI455X (gfx1250) — matrix operand order repaired and checked by reading, not compiled, not run on hardware
//
#include <hip/hip_runtime.h>
#include <hip/hip_bf16.h>
#include <math.h>

typedef __attribute__((ext_vector_type(16))) _Float16 v16h;
typedef __attribute__((ext_vector_type(8)))  float    v8f;

// Pointee types demanded by the gfx1250 async global->LDS copy builtins:
// payload as N x i32 vector, global (as1) source, LDS (as3) dest.
typedef __attribute__((vector_size(16))) int v4i_b128;
typedef __attribute__((address_space(1))) v4i_b128 g_v4i;
typedef __attribute__((address_space(3))) v4i_b128 l_v4i;

#define L_SEQ 128

__device__ __forceinline__ float siluf(float v) { return v / (1.0f + __expf(-v)); }

__device__ __forceinline__ void wait_async_all() {
#if __has_builtin(__builtin_amdgcn_s_wait_asynccnt)
  __builtin_amdgcn_s_wait_asynccnt(0);
#else
  asm volatile("s_wait_asynccnt 0" ::: "memory");
#endif
}

// Stage 4 consecutive floats (16 bytes) of the weight tile into LDS using the
// gfx1250 async global->LDS copy (GLOBAL_LOAD_ASYNC_TO_LDS_B128, ASYNCcnt).
__device__ __forceinline__ void stage_b128(float* lds_dst, const float* gsrc) {
#if __has_builtin(__builtin_amdgcn_global_load_async_to_lds_b128)
  __builtin_amdgcn_global_load_async_to_lds_b128(
      (g_v4i*)gsrc, (l_v4i*)lds_dst, 0, 0);
#else
  lds_dst[0] = gsrc[0];
  lds_dst[1] = gsrc[1];
  lds_dst[2] = gsrc[2];
  lds_dst[3] = gsrc[3];
#endif
}

// ---------------------------------------------------------------------------
// GEMM: Y[l, o] = sum_i X[l, i] * W[o, i]   (Y = X @ W^T), L fixed at 128.
// Block = 256 threads (8 waves); wave w computes M-tile w (rows 16w..16w+15)
// of a 16-wide N tile, so each weight element is pulled from HBM once per
// block. A 16x64 f32 weight tile (4 KB) is double-buffered in LDS and filled
// with b128 async global->LDS copies overlapped with two wmma sub-steps per
// barrier round. f32 operands are converted to f16 in-register; accumulation
// in f32 via V_WMMA_F32_16X16X32_F16. I must be a multiple of 64.
// ---------------------------------------------------------------------------
__global__ __launch_bounds__(256) void gemm_xwt_wmma(
    const float* __restrict__ X, int ldx,
    const float* __restrict__ W, int ldw,
    float* __restrict__ Y, int ldy, int I)
{
  __shared__ float Bt[2][16 * 64];   // [buf][n*64 + k]

  const int tid  = threadIdx.x;
  const int lane = tid & 31;
  const int wv   = tid >> 5;         // 0..7 -> M tile
  const int n16  = lane & 15;
  const int hi   = lane >> 4;
  const int nb   = blockIdx.x << 4;  // N tile base

  // Staging map: element e = 4*tid .. 4*tid+3: row n = tid>>4, col k = (tid&15)*4.
  const float* wstage = W + (size_t)(nb + (tid >> 4)) * ldw + (tid & 15) * 4;

  v8f acc = {};

  // Preload first K tile (16 rows x 64 K).
  stage_b128(&Bt[0][4 * tid], wstage);
  wait_async_all();
  __syncthreads();

  for (int k0 = 0; k0 < I; k0 += 64) {
    const int cur = (k0 >> 6) & 1;
    const int nxt = cur ^ 1;

    // Kick off async copy of the next weight tile before computing.
    if (k0 + 64 < I) stage_b128(&Bt[nxt][4 * tid], wstage + k0 + 64);

#pragma unroll
    for (int s = 0; s < 2; ++s) {
      // B fragment from LDS: lane n16 = N, K = s*32 + hi*16 + j
      v16h b;
#pragma unroll
      for (int j = 0; j < 16; ++j)
        b[j] = (_Float16)Bt[cur][n16 * 64 + s * 32 + hi * 16 + j];

      // A fragment from global (wave-private rows, L2-resident activations):
      // lane<16 -> M=n16, K = 0..15; lane>=16 -> K = 16..31
      const float* xrow = X + (size_t)(wv * 16 + n16) * ldx + k0 + s * 32 + hi * 16;
      __builtin_prefetch(xrow + 64, 0, 1);
      v16h a;
#pragma unroll
      for (int j = 0; j < 8; ++j) {
        a[j]     = (_Float16)xrow[j];
        a[j + 8] = (_Float16)xrow[8 + j];
      }

      acc = __builtin_amdgcn_wmma_f32_16x16x32_f16(
          false, a, false, b, (short)0, acc, false, false);
    }

    wait_async_all();
    __syncthreads();
  }

  // C/D layout: VGPR r -> M = r + hi*8, N = n16
#pragma unroll
  for (int r = 0; r < 8; ++r) {
    const int row = wv * 16 + r + hi * 8;
    Y[(size_t)row * ldy + nb + n16] = acc[r];
  }
}

// x[l,d] = pe[l,d] + condition * to_cond_w[d] + to_cond_b[d]   (d in [0,2048))
__global__ void build_x_kernel(const float* __restrict__ pe,
                               const float* __restrict__ cw,
                               const float* __restrict__ cb,
                               const float* __restrict__ cond,
                               float* __restrict__ x, int n)
{
  int i = blockIdx.x * blockDim.x + threadIdx.x;
  if (i >= n) return;
  int d = i & 2047;
  x[i] = pe[i] + cond[0] * cw[d] + cb[d];
}

// Depthwise causal conv (K=4) over L on xr = xz[:, :DI], then SiLU.
__global__ void conv_silu_kernel(const float* __restrict__ xz, int DI,
                                 const float* __restrict__ cw,
                                 const float* __restrict__ cb,
                                 float* __restrict__ xc, int n)
{
  int i = blockIdx.x * blockDim.x + threadIdx.x;
  if (i >= n) return;
  int l = i / DI;
  int d = i - l * DI;
  float s = cb[d];
#pragma unroll
  for (int t = 0; t < 4; ++t) {
    int ls = l + t - 3;
    if (ls >= 0) s += cw[d * 4 + t] * xz[(size_t)ls * (2 * DI) + d];
  }
  xc[i] = siluf(s);
}

// dt[l,d] = softplus(dtraw[l,d] + dt_b[d])   (in place)
__global__ void dt_act_kernel(float* __restrict__ dt,
                              const float* __restrict__ db, int DI, int n)
{
  int i = blockIdx.x * blockDim.x + threadIdx.x;
  if (i >= n) return;
  int d = i % DI;
  float v = dt[i] + db[d];
  dt[i] = (v > 20.0f) ? v : log1pf(__expf(v));
}

// Selective scan: one 32-thread block per channel d; lane owns n=lane, n=lane+32.
// h[l] = exp(dt*A)*h[l-1] + dt*B*xr;  y[l,d] = (sum_n h*C + xr*Dp) * silu(z)
__global__ __launch_bounds__(32) void scan_kernel(
    const float* __restrict__ dt, const float* __restrict__ xc,
    const float* __restrict__ dbc, int ldb, int dtr,
    const float* __restrict__ A_log, const float* __restrict__ Dp,
    const float* __restrict__ xz, int DI, float* __restrict__ y)
{
  const int d    = blockIdx.x;
  const int lane = threadIdx.x & 31;
  const float A1 = -__expf(A_log[(size_t)d * 64 + lane]);
  const float A2 = -__expf(A_log[(size_t)d * 64 + lane + 32]);
  const float* Bb = dbc + dtr;
  const float* Cb = dbc + dtr + 64;
  float h1 = 0.0f, h2 = 0.0f;

  for (int l = 0; l < L_SEQ; ++l) {
    float dtl = dt[(size_t)l * DI + d];
    float xrl = xc[(size_t)l * DI + d];
    float b1  = Bb[(size_t)l * ldb + lane];
    float b2  = Bb[(size_t)l * ldb + lane + 32];
    float c1  = Cb[(size_t)l * ldb + lane];
    float c2  = Cb[(size_t)l * ldb + lane + 32];
    float dbx = dtl * xrl;
    h1 = __expf(dtl * A1) * h1 + dbx * b1;
    h2 = __expf(dtl * A2) * h2 + dbx * b2;
    float p = h1 * c1 + h2 * c2;
#pragma unroll
    for (int off = 16; off > 0; off >>= 1) p += __shfl_xor(p, off, 32);
    if (lane == 0) {
      float zz = xz[(size_t)l * (2 * DI) + DI + d];
      y[(size_t)l * DI + d] = (p + xrl * Dp[d]) * siluf(zz);
    }
  }
}

extern "C" void kernel_launch(void* const* d_in, const int* in_sizes, int n_in,
                              void* d_out, int out_size, void* d_ws, size_t ws_size,
                              hipStream_t stream)
{
  (void)in_sizes; (void)n_in; (void)out_size; (void)ws_size;

  const float* pe        = (const float*)d_in[0];   // 128 x 2048
  const float* to_cond_w = (const float*)d_in[1];   // 2048
  const float* to_cond_b = (const float*)d_in[2];   // 2048
  const float* condition = (const float*)d_in[3];   // 1
  // d_in[4] = batch (int, ==1)
  const float* m1_in_proj  = (const float*)d_in[5];   // 8192 x 2048
  const float* m1_conv_w   = (const float*)d_in[6];   // 4096 x 4
  const float* m1_conv_b   = (const float*)d_in[7];   // 4096
  const float* m1_x_proj   = (const float*)d_in[8];   // 256 x 4096
  const float* m1_dt_w     = (const float*)d_in[9];   // 4096 x 128
  const float* m1_dt_b     = (const float*)d_in[10];  // 4096
  const float* m1_A_log    = (const float*)d_in[11];  // 4096 x 64
  const float* m1_D        = (const float*)d_in[12];  // 4096
  const float* m1_out_proj = (const float*)d_in[13];  // 2048 x 4096
  const float* m2_in_proj  = (const float*)d_in[14];  // 16384 x 2048
  const float* m2_conv_w   = (const float*)d_in[15];  // 8192 x 4
  const float* m2_conv_b   = (const float*)d_in[16];  // 8192
  const float* m2_x_proj   = (const float*)d_in[17];  // 384 x 8192
  const float* m2_dt_w     = (const float*)d_in[18];  // 8192 x 256
  const float* m2_dt_b     = (const float*)d_in[19];  // 8192
  const float* m2_A_log    = (const float*)d_in[20];  // 8192 x 64
  const float* m2_D        = (const float*)d_in[21];  // 8192
  const float* m2_out_proj = (const float*)d_in[22];  // 4096 x 8192

  float* out = (float*)d_out;                          // 128 x 4096
  float* ws  = (float*)d_ws;

  // Workspace layout (floats), total ~8.47M (~34 MB)
  float* x    = ws;               // 128*2048
  float* xz1  = x    +  262144;   // 128*8192
  float* xc1  = xz1  + 1048576;   // 128*4096
  float* dbc1 = xc1  +  524288;   // 128*256
  float* dt1  = dbc1 +   32768;   // 128*4096
  float* y1   = dt1  +  524288;   // 128*4096
  float* o1   = y1   +  524288;   // 128*2048
  float* xz2  = o1   +  262144;   // 128*16384
  float* xc2  = xz2  + 2097152;   // 128*8192
  float* dbc2 = xc2  + 1048576;   // 128*384
  float* dt2  = dbc2 +   49152;   // 128*8192
  float* y2   = dt2  + 1048576;   // 128*8192

  const int TB = 256;

  // ---- prelude: x = pe + cond*w + b ----
  build_x_kernel<<<(262144 + TB - 1) / TB, TB, 0, stream>>>(
      pe, to_cond_w, to_cond_b, condition, x, 262144);

  // ================= Mamba block 1 (DI=4096, dtr=128) =================
  gemm_xwt_wmma<<<8192 / 16, 256, 0, stream>>>(x, 2048, m1_in_proj, 2048, xz1, 8192, 2048);
  conv_silu_kernel<<<524288 / TB, TB, 0, stream>>>(xz1, 4096, m1_conv_w, m1_conv_b, xc1, 524288);
  gemm_xwt_wmma<<<256 / 16, 256, 0, stream>>>(xc1, 4096, m1_x_proj, 4096, dbc1, 256, 4096);
  gemm_xwt_wmma<<<4096 / 16, 256, 0, stream>>>(dbc1, 256, m1_dt_w, 128, dt1, 4096, 128);
  dt_act_kernel<<<524288 / TB, TB, 0, stream>>>(dt1, m1_dt_b, 4096, 524288);
  scan_kernel<<<4096, 32, 0, stream>>>(dt1, xc1, dbc1, 256, 128, m1_A_log, m1_D, xz1, 4096, y1);
  gemm_xwt_wmma<<<2048 / 16, 256, 0, stream>>>(y1, 4096, m1_out_proj, 4096, o1, 2048, 4096);

  // ================= Mamba block 2 (DI=8192, dtr=256) =================
  gemm_xwt_wmma<<<16384 / 16, 256, 0, stream>>>(o1, 2048, m2_in_proj, 2048, xz2, 16384, 2048);
  conv_silu_kernel<<<1048576 / TB, TB, 0, stream>>>(xz2, 8192, m2_conv_w, m2_conv_b, xc2, 1048576);
  gemm_xwt_wmma<<<384 / 16, 256, 0, stream>>>(xc2, 8192, m2_x_proj, 8192, dbc2, 384, 8192);
  gemm_xwt_wmma<<<8192 / 16, 256, 0, stream>>>(dbc2, 384, m2_dt_w, 256, dt2, 8192, 256);
  dt_act_kernel<<<1048576 / TB, TB, 0, stream>>>(dt2, m2_dt_b, 8192, 1048576);
  scan_kernel<<<8192, 32, 0, stream>>>(dt2, xc2, dbc2, 384, 256, m2_A_log, m2_D, xz2, 8192, y2);
  gemm_xwt_wmma<<<4096 / 16, 256, 0, stream>>>(y2, 8192, m2_out_proj, 8192, out, 4096, 8192);
}
